// ChildSumTreeLSTM_54537494725223
// MI455X (gfx1250) — compile-verified, matrix operand order checked by reading, not run on hardware
//
#include <hip/hip_runtime.h>
#include <hip/hip_bf16.h>
#include <stdint.h>

// ---------------------------------------------------------------------------
// Child-Sum Tree-LSTM for MI455X (gfx1250, wave32, WMMA bf16, TDM, 320KB LDS)
// ---------------------------------------------------------------------------

typedef __attribute__((ext_vector_type(16))) __bf16 v16bf;
typedef __attribute__((ext_vector_type(8)))  __bf16 v8bf;
typedef __attribute__((ext_vector_type(8)))  float  v8f;
typedef __attribute__((ext_vector_type(4)))  unsigned int u32x4;
typedef __attribute__((ext_vector_type(8)))  int    i32x8;
typedef __attribute__((ext_vector_type(4)))  int    i32x4;

union V16 { v16bf v; v8bf h[2]; };
union BF8 { uint4 u; __bf16 b[8]; };

// problem dims
constexpr int Bsz = 64, Nn = 512, Dm = 512, TD = 1536;
constexpr int ROWS = Bsz * Nn;          // 32768
constexpr int NWG  = 16;                // persistent workgroups for the scan

// ---- workspace layout (bytes) ----
constexpr size_t OFF_HSUM  = 0;                              // f32 [B][N][D]
constexpr size_t SZ_HSUM   = (size_t)Bsz * Nn * Dm * 4;      // 64 MiB
constexpr size_t OFF_FCSUM = OFF_HSUM + SZ_HSUM;             // f32 [B][N][D]
constexpr size_t OFF_BAR   = OFF_FCSUM + SZ_HSUM;            // 256 B barrier
constexpr size_t OFF_INBF  = OFF_BAR + 256;                  // bf16 [ROWS][512]
constexpr size_t SZ_INBF   = (size_t)ROWS * 512 * 2;
constexpr size_t OFF_WCAT  = OFF_INBF + SZ_INBF;             // bf16 swz [128][16][32][16]
constexpr size_t SZ_WCAT   = (size_t)2048 * 512 * 2;
constexpr size_t OFF_WIOUH = OFF_WCAT + SZ_WCAT;             // bf16 swz [32grp][3][16][32][16]
constexpr size_t SZ_WIOUH  = (size_t)512 * 1536 * 2;
constexpr size_t OFF_WFH   = OFF_WIOUH + SZ_WIOUH;           // bf16 swz [32grp][16][32][16]
constexpr size_t SZ_WFH    = (size_t)512 * 512 * 2;
constexpr size_t OFF_HRM   = OFF_WFH + SZ_WFH;               // bf16 [64][512] h staging
constexpr size_t OFF_IOUX  = OFF_HRM + 65536;                // f32 [ROWS][1536]
constexpr size_t SZ_IOUX   = (size_t)ROWS * TD * 4;
constexpr size_t OFF_FX    = OFF_IOUX + SZ_IOUX;             // f32 [ROWS][512]

// ---- LDS layout for the scan kernel (bytes) ----
constexpr int L_WI     = 0;                 // 2 groups * 48KB W_iouh slices
constexpr int L_WF     = 98304;             // 2 groups * 16KB W_fh slices
constexpr int L_A      = 131072;            // bf16 [64][520] staged A (padded stride)
constexpr int A_STRIDE = 520;               // 1040B rows -> conflict-free frag reads
constexpr int L_TS     = 197632;            // 8 waves * 272 floats transpose scratch
constexpr int SMEM_BYTES = 206336;

__device__ __forceinline__ float sigm(float x) { return 1.0f / (1.0f + __expf(-x)); }

__device__ __forceinline__ v8f wmma_bf16(v16bf a, v16bf b, v8f c) {
  return __builtin_amdgcn_wmma_f32_16x16x32_bf16(false, a, false, b, (short)0, c,
                                                 false, false);
}

// ---- Tensor Data Mover: contiguous global->LDS copy (ndw dwords) ----------
// 6-arg builtin variant: (g0 u32x4, g1 i32x8, g2 i32x4, g3 i32x4, g? i32x8, cpol)
__device__ __forceinline__ void tdm_copy_dwords(uint32_t lds_byte_off,
                                                const void* gsrc, uint32_t ndw) {
#if __has_builtin(__builtin_amdgcn_tensor_load_to_lds)
  uint64_t ga = (uint64_t)(uintptr_t)gsrc;
  u32x4 g0;
  g0[0] = 1u;                                             // count=1, user D#
  g0[1] = lds_byte_off;                                   // lds_addr
  g0[2] = (uint32_t)ga;                                   // global_addr[31:0]
  g0[3] = (uint32_t)((ga >> 32) & 0x01FFFFFFu) | (2u << 30); // addr[56:32], type=2
  i32x8 g1;
  g1[0] = (int)(2u << 16);                                // data_size = 4B dwords
  g1[1] = (int)((ndw & 0xFFFFu) << 16);                   // tensor_dim0[15:0]
  g1[2] = (int)(((ndw >> 16) & 0xFFFFu) | (1u << 16));    // dim0[31:16], tensor_dim1=1
  g1[3] = (int)((ndw & 0xFFFFu) << 16);                   // tile_dim0
  g1[4] = 1;                                              // tile_dim1 = 1
  g1[5] = (int)ndw;                                       // tensor_dim0_stride
  g1[6] = 0; g1[7] = 0;
  i32x4 z4; z4[0] = z4[1] = z4[2] = z4[3] = 0;
  i32x8 z8; z8[0] = z8[1] = z8[2] = z8[3] = z8[4] = z8[5] = z8[6] = z8[7] = 0;
  __builtin_amdgcn_tensor_load_to_lds(g0, g1, z4, z4, z8, 0);
#else
  (void)lds_byte_off; (void)gsrc; (void)ndw;
#endif
}

// ---- device-wide sense-reversing barrier for the persistent scan kernel ----
__device__ __forceinline__ void grid_barrier(unsigned* cnt, unsigned* gen) {
  __threadfence();
  __syncthreads();
  if (threadIdx.x == 0) {
    unsigned g = __hip_atomic_load(gen, __ATOMIC_SEQ_CST, __HIP_MEMORY_SCOPE_AGENT);
    if (__hip_atomic_fetch_add(cnt, 1u, __ATOMIC_SEQ_CST, __HIP_MEMORY_SCOPE_AGENT)
        == (unsigned)(NWG - 1)) {
      __hip_atomic_store(cnt, 0u, __ATOMIC_SEQ_CST, __HIP_MEMORY_SCOPE_AGENT);
      __hip_atomic_fetch_add(gen, 1u, __ATOMIC_SEQ_CST, __HIP_MEMORY_SCOPE_AGENT);
    } else {
      while (__hip_atomic_load(gen, __ATOMIC_SEQ_CST, __HIP_MEMORY_SCOPE_AGENT) == g)
        __builtin_amdgcn_s_sleep(8);
    }
  }
  __syncthreads();
}

// ===========================================================================
// Prep kernels
// ===========================================================================
__global__ void zero_kernel(float4* dst, size_t n4) {
  size_t t = (size_t)blockIdx.x * 256 + threadIdx.x;
  if (t < n4) dst[t] = make_float4(0.f, 0.f, 0.f, 0.f);
}

__global__ void cvt_inputs_kernel(const float* __restrict__ src,
                                  __bf16* __restrict__ dst, int n8) {
  int t = blockIdx.x * 256 + threadIdx.x;
  if (t >= n8) return;
  const float4* s = (const float4*)src + (size_t)t * 2;
  float4 x = s[0], y = s[1];
  BF8 p;
  p.b[0]=(__bf16)x.x; p.b[1]=(__bf16)x.y; p.b[2]=(__bf16)x.z; p.b[3]=(__bf16)x.w;
  p.b[4]=(__bf16)y.x; p.b[5]=(__bf16)y.y; p.b[6]=(__bf16)y.z; p.b[7]=(__bf16)y.w;
  *((uint4*)dst + t) = p.u;
}

// B-fragment swizzle convention (bf16 16x16x32 WMMA):
//   lane L: column = L%16, half = L/16; 16 values j=0..15 hold
//   K = kc*32 + (j<8 ? j + 8*half : 16 + (j-8) + 8*half)
// Each fragment is 32 lanes * 32B contiguous; A-operands use the mirrored
// layout (lane = row) and are read as two b128 chunks from row-major bf16.

__global__ void swizzle_wcat_kernel(const float* __restrict__ wioux,  // [512][1536]
                                    const float* __restrict__ wfx,    // [512][512]
                                    __bf16* __restrict__ dst) {
  int idx = blockIdx.x * 256 + threadIdx.x;          // frag-half id
  if (idx >= 128 * 16 * 32 * 2) return;
  int hj = idx & 1, lane = (idx >> 1) & 31, kc = (idx >> 6) & 15, nt = idx >> 10;
  int col = nt * 16 + (lane & 15);
  int k0  = kc * 32 + 8 * (lane >> 4) + hj * 16;
  BF8 p;
#pragma unroll
  for (int t = 0; t < 8; ++t) {
    int k = k0 + t;
    float v = (col < TD) ? wioux[(size_t)k * TD + col]
                         : wfx[(size_t)k * Dm + (col - TD)];
    p.b[t] = (__bf16)v;
  }
  *((uint4*)dst + idx) = p.u;
}

__global__ void swizzle_wiouh_kernel(const float* __restrict__ w,  // [512][1536]
                                     __bf16* __restrict__ dst) {
  int idx = blockIdx.x * 256 + threadIdx.x;          // 32*3*16*32*2 = 98304
  if (idx >= 32 * 3 * 16 * 32 * 2) return;
  int hj = idx & 1, lane = (idx >> 1) & 31, kc = (idx >> 6) & 15;
  int rest = idx >> 10;                              // grp*3 + chunk
  int chunk = rest % 3, grp = rest / 3;
  int col = chunk * 512 + grp * 16 + (lane & 15);
  int k0  = kc * 32 + 8 * (lane >> 4) + hj * 16;
  BF8 p;
#pragma unroll
  for (int t = 0; t < 8; ++t) p.b[t] = (__bf16)w[(size_t)(k0 + t) * TD + col];
  *((uint4*)dst + idx) = p.u;
}

__global__ void swizzle_wfh_kernel(const float* __restrict__ w,    // [512][512]
                                   __bf16* __restrict__ dst) {
  int idx = blockIdx.x * 256 + threadIdx.x;          // 32*16*32*2 = 32768
  if (idx >= 32 * 16 * 32 * 2) return;
  int hj = idx & 1, lane = (idx >> 1) & 31, kc = (idx >> 6) & 15, grp = idx >> 10;
  int col = grp * 16 + (lane & 15);
  int k0  = kc * 32 + 8 * (lane >> 4) + hj * 16;
  BF8 p;
#pragma unroll
  for (int t = 0; t < 8; ++t) p.b[t] = (__bf16)w[(size_t)(k0 + t) * Dm + col];
  *((uint4*)dst + idx) = p.u;
}

// ===========================================================================
// Big input projection: [32768x512] @ [512x2048] -> ioux_all | fx_all (f32)
// Each wave: one 16-row M-tile x four 16-col N-tiles, K=512 (16 WMMA chunks),
// software-pipelined with a 2-deep register double buffer so global_load_b128
// latency overlaps WMMA issue (partial s_wait_loadcnt instead of 0).
// ===========================================================================
__global__ void __launch_bounds__(256)
input_gemm_kernel(const __bf16* __restrict__ inbf,   // [ROWS][512] row-major
                  const __bf16* __restrict__ wcat,   // swizzled B fragments
                  const float* __restrict__ b_ioux,
                  const float* __restrict__ b_fx,
                  float* __restrict__ ioux,
                  float* __restrict__ fx) {
  const int lane = threadIdx.x & 31;
  const int wv   = threadIdx.x >> 5;
  const int task = blockIdx.x * 8 + wv;     // 0..65535
  const int mt   = task >> 5;               // 0..2047  (M tile)
  const int ntg  = task & 31;               // 0..31    (group of 4 N tiles)
  const int nl   = lane & 15, half = lane >> 4;

  const __bf16* arow = inbf + (size_t)(mt * 16 + nl) * 512 + 8 * half;
  const char*   wbase = (const char*)wcat +
                        ((size_t)(ntg * 4) * 16 * 32 + lane) * 32;

  const v8f zf = {0.f, 0.f, 0.f, 0.f, 0.f, 0.f, 0.f, 0.f};
  v8f acc[4] = {zf, zf, zf, zf};

  V16   abuf[2];
  v16bf wbuf[2][4];

  // prologue: fetch kc = 0
  abuf[0].h[0] = *(const v8bf*)(arow);
  abuf[0].h[1] = *(const v8bf*)(arow + 16);
#pragma unroll
  for (int t = 0; t < 4; ++t)
    wbuf[0][t] = *(const v16bf*)(wbase + (size_t)t * 16 * 32 * 32);

#pragma unroll
  for (int kc = 0; kc < 16; ++kc) {
    const int cur = kc & 1, nxt = cur ^ 1;
    if (kc < 15) {   // issue next K-chunk's loads before consuming current
      const __bf16* ap = arow + (kc + 1) * 32;
      abuf[nxt].h[0] = *(const v8bf*)ap;
      abuf[nxt].h[1] = *(const v8bf*)(ap + 16);
#pragma unroll
      for (int t = 0; t < 4; ++t)
        wbuf[nxt][t] = *(const v16bf*)(wbase + ((size_t)t * 16 * 32 * 32) +
                                       (size_t)(kc + 1) * 32 * 32);
    }
#pragma unroll
    for (int t = 0; t < 4; ++t)
      acc[t] = wmma_bf16(abuf[cur].v, wbuf[cur][t], acc[t]);
  }

#pragma unroll
  for (int t = 0; t < 4; ++t) {
    int col = (ntg * 4 + t) * 16 + nl;
    float bias = (col < TD) ? b_ioux[col] : b_fx[col - TD];
#pragma unroll
    for (int r = 0; r < 8; ++r) {
      int row = mt * 16 + 8 * half + r;
      float v = acc[t][r] + bias;
      if (col < TD) ioux[(size_t)row * TD + col] = v;
      else          fx[(size_t)row * Dm + (col - TD)] = v;
    }
  }
}

// ===========================================================================
// Persistent tree-scan kernel: 16 WGs x 256 threads (8 waves).
// WG owns 2 column-groups of 16; wave = (group, M-tile). Recurrent weights
// are LDS-resident for all 512 steps (TDM preload). 2 grid barriers/step.
// ===========================================================================
__global__ void __launch_bounds__(256, 1)
tree_scan_kernel(const float* __restrict__ ioux,     // [ROWS][1536]
                 const float* __restrict__ fx,       // [ROWS][512]
                 const float* __restrict__ b_iouh,
                 const float* __restrict__ b_fh,
                 const int*   __restrict__ parents,  // [B][N]
                 float* __restrict__ h_sum,          // [B][N][D]
                 float* __restrict__ fc_sum,         // [B][N][D]
                 const __bf16* __restrict__ wiouh_swz,
                 const __bf16* __restrict__ wfh_swz,
                 __bf16* __restrict__ h_rm,          // [64][512] bf16 staging
                 unsigned* __restrict__ bar,
                 float* __restrict__ out) {          // h_all | c_root | h_root
  extern __shared__ char smem[];
  const int tid  = threadIdx.x;
  const int lane = tid & 31;
  const int wv   = tid >> 5;            // 0..7
  const int gl   = wv >> 2;             // local group 0..1
  const int mt   = wv & 3;              // M tile 0..3 (rows of batch)
  const int grp  = blockIdx.x * 2 + gl; // global column group 0..31
  const int nl   = lane & 15;
  const int half = lane >> 4;
  const int colg = grp * 16 + nl;

  __bf16* Ald = (__bf16*)(smem + L_A);
  float*  tsc = (float*)(smem + L_TS) + wv * 272;
  const v8f zf = {0.f, 0.f, 0.f, 0.f, 0.f, 0.f, 0.f, 0.f};

  // ---- one-time weight preload into LDS via the Tensor Data Mover ----
  uint32_t lbase = (uint32_t)(uintptr_t)smem;
#if __has_builtin(__builtin_amdgcn_tensor_load_to_lds)
  if (wv == 0) {
    tdm_copy_dwords(lbase + L_WI,
                    (const char*)wiouh_swz + (size_t)blockIdx.x * 98304, 98304 / 4);
    tdm_copy_dwords(lbase + L_WF,
                    (const char*)wfh_swz + (size_t)blockIdx.x * 32768, 32768 / 4);
#if __has_builtin(__builtin_amdgcn_s_wait_tensorcnt)
    __builtin_amdgcn_s_wait_tensorcnt(0);
#endif
  }
#else
  {
    const uint4* s0 = (const uint4*)((const char*)wiouh_swz + (size_t)blockIdx.x * 98304);
    uint4* d0 = (uint4*)(smem + L_WI);
    for (int t = tid; t < 98304 / 16; t += 256) d0[t] = s0[t];
    const uint4* s1 = (const uint4*)((const char*)wfh_swz + (size_t)blockIdx.x * 32768);
    uint4* d1 = (uint4*)(smem + L_WF);
    for (int t = tid; t < 32768 / 16; t += 256) d1[t] = s1[t];
  }
#endif
  __syncthreads();

  const float bi = b_iouh[colg];
  const float bo = b_iouh[512 + colg];
  const float bu = b_iouh[1024 + colg];
  const float bfv = b_fh[colg];

  for (int i = Nn - 1; i >= 0; --i) {
    // ---- P0: stage A = bf16(h_sum[:, i, :]) into LDS row-major ----
    for (int t = tid; t < (Bsz * Dm) / 8; t += 256) {
      int base = t * 8;
      int b = base >> 9;
      int col = base & 511;
      const float* s = h_sum + ((size_t)(b * Nn + i) << 9) + col;
      float4 x = *(const float4*)s;
      float4 y = *(const float4*)(s + 4);
      BF8 p;
      p.b[0]=(__bf16)x.x; p.b[1]=(__bf16)x.y; p.b[2]=(__bf16)x.z; p.b[3]=(__bf16)x.w;
      p.b[4]=(__bf16)y.x; p.b[5]=(__bf16)y.y; p.b[6]=(__bf16)y.z; p.b[7]=(__bf16)y.w;
      *(uint4*)(Ald + b * A_STRIDE + col) = p.u;
    }
    __syncthreads();

    // ---- P1: iou gates = A @ W_iouh (3 chunks for this wave's 16 cols) ----
    v8f ai = zf, ao = zf, au = zf;
    const char* wiB = smem + L_WI + gl * 49152;
#pragma unroll 4
    for (int kc = 0; kc < 16; ++kc) {
      V16 a;
      const __bf16* ap = Ald + (mt * 16 + nl) * A_STRIDE + kc * 32 + 8 * half;
      a.h[0] = *(const v8bf*)ap;
      a.h[1] = *(const v8bf*)(ap + 16);
      v16bf w0 = *(const v16bf*)(wiB + ((size_t)((0 * 16 + kc) * 32 + lane)) * 32);
      v16bf w1 = *(const v16bf*)(wiB + ((size_t)((1 * 16 + kc) * 32 + lane)) * 32);
      v16bf w2 = *(const v16bf*)(wiB + ((size_t)((2 * 16 + kc) * 32 + lane)) * 32);
      ai = wmma_bf16(a.v, w0, ai);
      ao = wmma_bf16(a.v, w1, ao);
      au = wmma_bf16(a.v, w2, au);
    }

    float hreg[8], creg[8];
#pragma unroll
    for (int r = 0; r < 8; ++r) {
      int b = mt * 16 + 8 * half + r;
      size_t rowi = (size_t)(b * Nn + i);
      const float* iox = ioux + rowi * TD;
      float ig = sigm(ai[r] + iox[colg] + bi);
      float og = sigm(ao[r] + iox[512 + colg] + bo);
      float ug = sigm(au[r] + iox[1024 + colg] + bu);   // note: u uses sigmoid
      float cc = ig * ug + fc_sum[(rowi << 9) + colg];
      float hh = og * tanhf(cc);
      creg[r] = cc; hreg[r] = hh;
      out[(rowi << 9) + colg] = hh;                     // o_states
      if (i == 0) {
        out[(size_t)ROWS * Dm + (size_t)b * Dm + colg] = cc;                       // c_root
        out[(size_t)ROWS * Dm + (size_t)Bsz * Dm + (size_t)b * Dm + colg] = hh;    // h_root
      }
    }

    // ---- transpose h (accumulator layout -> row-major bf16 staging) ----
#pragma unroll
    for (int r = 0; r < 8; ++r) tsc[(8 * half + r) * 17 + nl] = hreg[r];
    __syncthreads();
    {
      BF8 p;
#pragma unroll
      for (int t = 0; t < 8; ++t) p.b[t] = (__bf16)tsc[nl * 17 + 8 * half + t];
      *(uint4*)(h_rm + (size_t)(mt * 16 + nl) * 512 + grp * 16 + 8 * half) = p.u;
    }
    grid_barrier(bar, bar + 16);   // all h staged before f-gate GEMM

    // ---- P2: f = sigmoid(h @ W_fh + b_fh + fx[parent]); scatter-add ----
    v8f af = zf;
    const char* wfB = smem + L_WF + gl * 16384;
#pragma unroll 4
    for (int kc = 0; kc < 16; ++kc) {
      V16 a;
      const __bf16* ap = h_rm + (size_t)(mt * 16 + nl) * 512 + kc * 32 + 8 * half;
      a.h[0] = *(const v8bf*)ap;
      a.h[1] = *(const v8bf*)(ap + 16);
      v16bf w = *(const v16bf*)(wfB + ((size_t)(kc * 32 + lane)) * 32);
      af = wmma_bf16(a.v, w, af);
    }
#pragma unroll
    for (int r = 0; r < 8; ++r) {
      int b = mt * 16 + 8 * half + r;
      int p = parents[b * Nn + i];
      if (p >= 0) {
        size_t rp = (size_t)(b * Nn + p);
        float f = sigm(af[r] + bfv + fx[(rp << 9) + colg]);
        size_t o = (rp << 9) + colg;                 // unique per lane this step
        h_sum[o]  += hreg[r];
        fc_sum[o] += f * creg[r];
      }
    }
    grid_barrier(bar, bar + 16);   // scatter visible before next step's P0
  }
}

// ===========================================================================
extern "C" void kernel_launch(void* const* d_in, const int* in_sizes, int n_in,
                              void* d_out, int out_size, void* d_ws, size_t ws_size,
                              hipStream_t stream) {
  (void)in_sizes; (void)n_in; (void)out_size; (void)ws_size;
  const float* inputs  = (const float*)d_in[0];
  const int*   parents = (const int*)d_in[1];
  const float* W_ioux  = (const float*)d_in[2];
  const float* b_ioux  = (const float*)d_in[3];
  const float* W_iouh  = (const float*)d_in[4];
  const float* b_iouh  = (const float*)d_in[5];
  const float* W_fx    = (const float*)d_in[6];
  const float* b_fx    = (const float*)d_in[7];
  const float* W_fh    = (const float*)d_in[8];
  const float* b_fh    = (const float*)d_in[9];

  char* ws = (char*)d_ws;
  float*    h_sum     = (float*)(ws + OFF_HSUM);
  float*    fc_sum    = (float*)(ws + OFF_FCSUM);
  unsigned* bar       = (unsigned*)(ws + OFF_BAR);
  __bf16*   inbf      = (__bf16*)(ws + OFF_INBF);
  __bf16*   wcat      = (__bf16*)(ws + OFF_WCAT);
  __bf16*   wiouh_swz = (__bf16*)(ws + OFF_WIOUH);
  __bf16*   wfh_swz   = (__bf16*)(ws + OFF_WFH);
  __bf16*   h_rm      = (__bf16*)(ws + OFF_HRM);
  float*    ioux      = (float*)(ws + OFF_IOUX);
  float*    fx        = (float*)(ws + OFF_FX);

  (void)hipFuncSetAttribute((const void*)tree_scan_kernel,
                            hipFuncAttributeMaxDynamicSharedMemorySize, SMEM_BYTES);

  // zero h_sum + fc_sum + barrier (covers OFF_HSUM..OFF_BAR+256)
  size_t n4 = (SZ_HSUM * 2 + 256) / 16;
  zero_kernel<<<(unsigned)((n4 + 255) / 256), 256, 0, stream>>>((float4*)ws, n4);

  cvt_inputs_kernel<<<8192, 256, 0, stream>>>(inputs, inbf, ROWS * 512 / 8);
  swizzle_wcat_kernel <<<512, 256, 0, stream>>>(W_ioux, W_fx, wcat);
  swizzle_wiouh_kernel<<<384, 256, 0, stream>>>(W_iouh, wiouh_swz);
  swizzle_wfh_kernel  <<<128, 256, 0, stream>>>(W_fh, wfh_swz);

  input_gemm_kernel<<<8192, 256, 0, stream>>>(inbf, wcat, b_ioux, b_fx, ioux, fx);

  tree_scan_kernel<<<NWG, 256, SMEM_BYTES, stream>>>(
      ioux, fx, b_iouh, b_fh, parents, h_sum, fc_sum,
      wiouh_swz, wfh_swz, h_rm, bar, (float*)d_out);
}
